// DOC_62474594287887
// MI455X (gfx1250) — matrix operand order repaired and checked by reading, not compiled, not run on hardware
//
#include <hip/hip_runtime.h>
#include <hip/hip_bf16.h>
#include <stdint.h>

// ---------------------------------------------------------------------------
// DOC contrastive loss for MI455X (gfx1250, wave32, WMMA + async-to-LDS).
//   inputs:  current_preds f32[8,1,64,64], history_preds f32[8,1,64,64],
//            features f32[8,256,64,64], epoch (ignored, epoch>=1 path)
//   outputs: labels f32[8,1,64,64] ++ loss f32[1]   (32769 floats)
// Hot path: per-frame 4096x4096 Gram of 256-dim L2-normalized features via
// v_wmma_f32_16x16x32_f16, B panels double-buffered in LDS with
// global_load_async_to_lds_b128 (ASYNCcnt), fused exp + bg-masked row sum.
// ---------------------------------------------------------------------------

typedef __attribute__((ext_vector_type(16))) _Float16 v16h;
typedef __attribute__((ext_vector_type(8)))  _Float16 v8h;
typedef __attribute__((ext_vector_type(8)))  float    v8f;

#define T_FRAMES 8
#define C_DIM    256
#define NPIX     4096            // 64*64
#define INV_TEMP (1.0f / 0.07f)
#define EPS_LOSS 1e-8f
#define NORM_EPS 1e-12f

// ---------------------------------------------------------------------------
// Kernel 1: supervised labels (Dice-like deviation select) + bg mask + counts
// ---------------------------------------------------------------------------
__global__ __launch_bounds__(256)
void doc_labels_kernel(const float* __restrict__ cur,
                       const float* __restrict__ hist,
                       float* __restrict__ out_labels,
                       float* __restrict__ bgf,
                       float* __restrict__ nfg,
                       float* __restrict__ validf)
{
    const int t   = blockIdx.x;
    const int tid = threadIdx.x;
    const int base = t * NPIX;

    __shared__ float red[256];
    __shared__ int   ucs;

    float s1 = 0.f, s2 = 0.f;
    #pragma unroll
    for (int i = 0; i < 16; ++i) {
        int p = tid + i * 256;
        float c = cur[base + p];
        float h = hist[base + p];
        float cb = (c > 0.5f) ? 1.f : 0.f;
        float hb = (h > 0.5f) ? 1.f : 0.f;
        s1 += cb * hb;
        s2 += cb + hb;
    }

    red[tid] = s1; __syncthreads();
    for (int off = 128; off > 0; off >>= 1) {
        if (tid < off) red[tid] += red[tid + off];
        __syncthreads();
    }
    const float e1 = red[0];
    __syncthreads();

    red[tid] = s2; __syncthreads();
    for (int off = 128; off > 0; off >>= 1) {
        if (tid < off) red[tid] += red[tid + off];
        __syncthreads();
    }
    if (tid == 0) {
        float e2 = red[0];
        float m1 = (2.f * e1 + EPS_LOSS) / (e2 + EPS_LOSS);
        float m2 = (e1 + EPS_LOSS) / (e2 - e1 + EPS_LOSS);
        float dev = 1.f - 0.5f * (m1 + m2);
        ucs = (dev <= 0.0f) ? 1 : 0;   // THRESH = 0.0
    }
    __syncthreads();
    const int uc = ucs;

    float sfg = 0.f;
    #pragma unroll
    for (int i = 0; i < 16; ++i) {
        int p = tid + i * 256;
        float c = cur[base + p];
        float h = hist[base + p];
        float lbl = uc ? c : h;
        out_labels[base + p] = lbl;
        float fg = (lbl > 0.5f) ? 1.f : 0.f;
        bgf[base + p] = 1.f - fg;
        sfg += fg;
    }
    red[tid] = sfg; __syncthreads();
    for (int off = 128; off > 0; off >>= 1) {
        if (tid < off) red[tid] += red[tid + off];
        __syncthreads();
    }
    if (tid == 0) {
        float n = red[0];
        nfg[t] = n;
        validf[t] = (n > 0.5f && n < (float)NPIX - 0.5f) ? 1.f : 0.f;
    }
}

// ---------------------------------------------------------------------------
// Kernel 2: per-pixel inverse L2 norm of 256-dim feature vectors
// ---------------------------------------------------------------------------
__global__ __launch_bounds__(256)
void doc_invnorm_kernel(const float* __restrict__ feat,
                        float* __restrict__ nrm2,
                        float* __restrict__ invn)
{
    const int p = blockIdx.x * 256 + threadIdx.x;   // [0, T*NPIX)
    const int t = p >> 12;
    const int n = p & (NPIX - 1);
    const float* fp = feat + (size_t)t * C_DIM * NPIX + n;
    float ss = 0.f;
    #pragma unroll 8
    for (int c = 0; c < C_DIM; ++c) {
        float v = fp[(size_t)c * NPIX];
        ss += v * v;
    }
    float nrm = sqrtf(ss);
    nrm2[p] = ss;
    invn[p] = 1.0f / fmaxf(nrm, NORM_EPS);
}

// ---------------------------------------------------------------------------
// Kernel 3: transpose [C][N] -> [N][C], scale by invn, convert to f16
// ---------------------------------------------------------------------------
__global__ __launch_bounds__(256)
void doc_transpose_f16_kernel(const float* __restrict__ feat,
                              const float* __restrict__ invn,
                              _Float16* __restrict__ featH)
{
    const int t  = blockIdx.z;
    const int n0 = blockIdx.x * 32;
    const int c0 = blockIdx.y * 32;
    const int tx = threadIdx.x;      // 0..31
    const int ty = threadIdx.y;      // 0..7

    __shared__ float tile[32][33];

    const float* fbase = feat + (size_t)t * C_DIM * NPIX;
    #pragma unroll
    for (int i = 0; i < 4; ++i) {
        int c = c0 + ty + i * 8;
        tile[ty + i * 8][tx] = fbase[(size_t)c * NPIX + n0 + tx];
    }
    __syncthreads();
    #pragma unroll
    for (int i = 0; i < 4; ++i) {
        int n = n0 + ty + i * 8;
        float s = invn[t * NPIX + n];
        float v = tile[tx][ty + i * 8] * s;
        featH[((size_t)t * NPIX + n) * C_DIM + c0 + tx] = (_Float16)v;
    }
}

// ---------------------------------------------------------------------------
// Kernel 4: Gram matrix via WMMA f16 + async-to-LDS double-buffered B panels.
// grid = (32 row-blocks, 8 frames); block = 256 threads = 8 waves.
// Each wave owns a distinct 16-row A strip (rbase = blk*128 + wave*16);
// all waves share B column panels (16 cols x 256 f16 = 8 KB, contiguous in
// the [T][N][C] layout), staged with global_load_async_to_lds_b128.
//
// Dynamic LDS layout (bytes):
//   [0,      8192)  B panel buffer 0
//   [8192,  16384)  B panel buffer 1
//   [16384, 32768)  bg mask for the frame (4096 floats)
// ---------------------------------------------------------------------------
__global__ __launch_bounds__(256)
void doc_gram_neg_kernel(const _Float16* __restrict__ featH,
                         const float* __restrict__ bgf,
                         float* __restrict__ neg)
{
    const int t     = blockIdx.y;
    const int tid   = threadIdx.x;
    const int lane  = tid & 31;
    const int wave  = tid >> 5;
    const int l16   = lane & 15;
    const int h     = lane >> 4;
    const int rbase = blockIdx.x * 128 + wave * 16;

    extern __shared__ char smem_raw[];
    _Float16* smemB  = (_Float16*)smem_raw;            // 2 x 4096 halfs
    float*    smemBg = (float*)(smem_raw + 16384);     // 4096 floats

    // LDS byte base for inline-asm async destinations (dynamic LDS starts
    // right after static LDS).
    const uint32_t ldsDynBase = __builtin_amdgcn_groupstaticsize();

    // stage bg mask for the whole frame into LDS (plain loads, pre-loop)
    #pragma unroll
    for (int i = 0; i < 16; ++i)
        smemBg[tid + i * 256] = bgf[t * NPIX + tid + i * 256];

    // Preload all 8 K-step A fragments for this wave's 16-row strip.
    // A frag (16x32 f16, lane l: row=l%16, h=l/16): K chunk {16h..16h+15}
    // -> one contiguous 32B load.
    const _Float16* fr = featH + ((size_t)t * NPIX + rbase + l16) * C_DIM;
    v16h aF[8];
    #pragma unroll
    for (int ks = 0; ks < 8; ++ks) {
        const int kb = ks * 32;
        aF[ks] = *(const v16h*)(fr + kb + 16 * h);
    }

    const _Float16* frameB = featH + (size_t)t * NPIX * C_DIM;

    // async panel issue: 8 KB contiguous; 256 threads x 16B, two halves.
    // Per ISA 15.18: INST_OFFSET is added to BOTH the LDS and global address.
    auto issue_panel = [&](int ct, int buf) {
        uint64_t g = (uint64_t)(const void*)(frameB + (size_t)ct * 16 * C_DIM)
                   + (uint32_t)(tid * 16);
        uint32_t l = ldsDynBase + (uint32_t)(buf * 8192 + tid * 16);
        asm volatile("global_load_async_to_lds_b128 %0, %1, off"
                     :: "v"(l), "v"(g) : "memory");
        asm volatile("global_load_async_to_lds_b128 %0, %1, off offset:4096"
                     :: "v"(l), "v"(g) : "memory");
    };

    float negAcc[8];
    #pragma unroll
    for (int j = 0; j < 8; ++j) negAcc[j] = 0.f;

    issue_panel(0, 0);

    for (int ct = 0; ct < NPIX / 16; ++ct) {
        if (ct + 1 < NPIX / 16) {
            issue_panel(ct + 1, (ct + 1) & 1);
            // in-order async completion: <=2 outstanding means panel ct landed
            asm volatile("s_wait_asynccnt 2" ::: "memory");
        } else {
            asm volatile("s_wait_asynccnt 0" ::: "memory");
        }
        __syncthreads();   // all waves' portions of panel ct are in LDS

        // B frag (32x16 f16, lane l: col=l%16, h=l/16): K chunk {16h..16h+15}
        const _Float16* bp = smemB + (ct & 1) * 4096 + l16 * C_DIM + 16 * h;
        v16h bF[8];
        #pragma unroll
        for (int ks = 0; ks < 8; ++ks)
            bF[ks] = *(const v16h*)(bp + ks * 32);

        v8f c = {};
        #pragma unroll
        for (int ks = 0; ks < 8; ++ks)
            c = __builtin_amdgcn_wmma_f32_16x16x32_f16(
                    false, aF[ks], false, bF[ks], (short)0, c, false, false);

        // fused epilogue: exp(sim/TEMP) masked by bg(column), row-accumulate
        const float bgv = smemBg[ct * 16 + l16];
        #pragma unroll
        for (int j = 0; j < 8; ++j)
            negAcc[j] += bgv * __expf(c[j] * INV_TEMP);

        __syncthreads();   // done reading this buffer; next iter may overwrite
    }

    // Row j+8h lives in the 16 lanes of half h: butterfly-reduce across them.
    #pragma unroll
    for (int j = 0; j < 8; ++j) {
        float v = negAcc[j];
        v += __shfl_xor(v, 1, 32);
        v += __shfl_xor(v, 2, 32);
        v += __shfl_xor(v, 4, 32);
        v += __shfl_xor(v, 8, 32);
        negAcc[j] = v;
    }
    if (l16 == 0) {
        #pragma unroll
        for (int j = 0; j < 8; ++j)
            neg[(size_t)t * NPIX + rbase + 8 * h + j] = negAcc[j];
    }
}

// ---------------------------------------------------------------------------
// Kernel 5a: per-frame fg-mean of per-pixel loss
// ---------------------------------------------------------------------------
__global__ __launch_bounds__(256)
void doc_frame_loss_kernel(const float* __restrict__ neg,
                           const float* __restrict__ bgf,
                           const float* __restrict__ nrm2,
                           const float* __restrict__ invn,
                           const float* __restrict__ nfg,
                           const float* __restrict__ validf,
                           float* __restrict__ fl)
{
    const int t   = blockIdx.x;
    const int tid = threadIdx.x;
    const int base = t * NPIX;
    __shared__ float red[256];

    float acc = 0.f;
    #pragma unroll
    for (int i = 0; i < 16; ++i) {
        int p = base + tid + i * 256;
        float fg = 1.f - bgf[p];
        float in = invn[p];
        float d  = nrm2[p] * in * in;             // ||f_hat||^2 (=1 unless tiny)
        float pos = __expf(d * INV_TEMP);
        float nv  = neg[p];
        float lp  = -__logf(pos / (pos + nv + EPS_LOSS));
        acc += fg * lp;
    }
    red[tid] = acc; __syncthreads();
    for (int off = 128; off > 0; off >>= 1) {
        if (tid < off) red[tid] += red[tid + off];
        __syncthreads();
    }
    if (tid == 0)
        fl[t] = (red[0] / fmaxf(nfg[t], 1.f)) * validf[t];
}

// ---------------------------------------------------------------------------
// Kernel 5b: final scalar
// ---------------------------------------------------------------------------
__global__ void doc_final_kernel(const float* __restrict__ fl,
                                 const float* __restrict__ validf,
                                 float* __restrict__ out_loss)
{
    if (threadIdx.x == 0 && blockIdx.x == 0) {
        float s = 0.f, nv = 0.f;
        #pragma unroll
        for (int t = 0; t < T_FRAMES; ++t) { s += fl[t]; nv += validf[t]; }
        out_loss[0] = (nv > 0.f) ? (s / fmaxf(nv, 1.f)) : 0.f;
    }
}

// ---------------------------------------------------------------------------
extern "C" void kernel_launch(void* const* d_in, const int* in_sizes, int n_in,
                              void* d_out, int out_size, void* d_ws, size_t ws_size,
                              hipStream_t stream)
{
    const float* cur  = (const float*)d_in[0];   // [8,1,64,64]
    const float* hist = (const float*)d_in[1];   // [8,1,64,64]
    const float* feat = (const float*)d_in[2];   // [8,256,64,64]
    // d_in[3] = epoch, ignored (epoch >= 1 path)

    float* out = (float*)d_out;                  // labels[32768] ++ loss[1]
    float* out_labels = out;
    float* out_loss   = out + T_FRAMES * NPIX;

    // workspace layout
    char* ws = (char*)d_ws;
    const size_t FEATH_BYTES = (size_t)T_FRAMES * NPIX * C_DIM * sizeof(_Float16); // 16 MB
    _Float16* featH = (_Float16*)ws;
    float* invn   = (float*)(ws + FEATH_BYTES);
    float* nrm2   = invn   + T_FRAMES * NPIX;
    float* bgf    = nrm2   + T_FRAMES * NPIX;
    float* neg    = bgf    + T_FRAMES * NPIX;
    float* nfg    = neg    + T_FRAMES * NPIX;
    float* validf = nfg    + T_FRAMES;
    float* fl     = validf + T_FRAMES;

    // 1) labels + bg mask + fg counts
    doc_labels_kernel<<<dim3(T_FRAMES), dim3(256), 0, stream>>>(
        cur, hist, out_labels, bgf, nfg, validf);

    // 2) inverse norms
    doc_invnorm_kernel<<<dim3(T_FRAMES * NPIX / 256), dim3(256), 0, stream>>>(
        feat, nrm2, invn);

    // 3) normalize + transpose + f16 convert
    doc_transpose_f16_kernel<<<dim3(NPIX / 32, C_DIM / 32, T_FRAMES),
                               dim3(32, 8), 0, stream>>>(feat, invn, featH);

    // 4) WMMA Gram + fused exp / masked row-sum (32 KB dynamic LDS)
    doc_gram_neg_kernel<<<dim3(NPIX / 128, T_FRAMES), dim3(256), 32768, stream>>>(
        featH, bgf, neg);

    // 5) per-frame fg-mean + final scalar
    doc_frame_loss_kernel<<<dim3(T_FRAMES), dim3(256), 0, stream>>>(
        neg, bgf, nrm2, invn, nfg, validf, fl);
    doc_final_kernel<<<dim3(1), dim3(32), 0, stream>>>(fl, validf, out_loss);
}
